// Transformer_6631429505602
// MI455X (gfx1250) — matrix operand order repaired and checked by reading, not compiled, not run on hardware
//
#include <hip/hip_runtime.h>
#include <hip/hip_bf16.h>

typedef __bf16 bf16_t;
typedef __attribute__((ext_vector_type(8)))  __bf16 v8bf;
typedef __attribute__((ext_vector_type(16))) __bf16 v16bf;
typedef __attribute__((ext_vector_type(8)))  float  v8f;

#define BB   8
#define TT   1024
#define DD   512
#define DFF  2048
#define HH   8
#define DHH  64
#define NEG_BIG_F (-4294967295.0f)
#define PE_SCALE  22.62741699796952f

// XOR swizzle for the 16x1024 LDS score strip: permutes within a row, keeps
// 8-float groups contiguous, spreads rows across banks (16-way -> 2-way).
#define SWZ(m, n) ((n) ^ (((m) & 7) << 3))

// ---------------------------------------------------------------------------
// WMMA fragment helpers (layouts per CDNA5 ISA 7.12.2, wave32)
// A 16x32 bf16: lane L: row M=L&15, K chunks at [(L>>4)*8 ..+7] and [16+(L>>4)*8 ..+7]
// B 32x16 bf16: lane L: col N=L&15, K chunks at [(L>>4)*8 ..+7] and [16+(L>>4)*8 ..+7] (same as A)
// C/D 16x16 f32: vgpr r, lane L: M = r + (L>>4)*8, N = L&15
// ---------------------------------------------------------------------------
__device__ __forceinline__ v16bf load_frag_a(const bf16_t* __restrict__ base, int stride) {
  const int lane = threadIdx.x & 31;
  const int row  = lane & 15;
  const int koff = (lane >> 4) * 8;
  const bf16_t* p0 = base + (size_t)row * stride + koff;
  v8bf lo = *(const v8bf*)p0;
  v8bf hi = *(const v8bf*)(p0 + 16);
  v16bf r;
#pragma unroll
  for (int i = 0; i < 8; ++i) { r[i] = lo[i]; r[8 + i] = hi[i]; }
  return r;
}

__device__ __forceinline__ v16bf load_frag_b(const bf16_t* __restrict__ base, int stride) {
  const int lane = threadIdx.x & 31;
  const int col  = lane & 15;
  const int koff = (lane >> 4) * 8;
  const bf16_t* p = base + (size_t)col * stride + koff;
  v8bf lo = *(const v8bf*)p;
  v8bf hi = *(const v8bf*)(p + 16);
  v16bf r;
#pragma unroll
  for (int i = 0; i < 8; ++i) { r[i] = lo[i]; r[8 + i] = hi[i]; }
  return r;
}

__device__ __forceinline__ v8f wmma_bf16(v16bf a, v16bf b, v8f c) {
  return __builtin_amdgcn_wmma_f32_16x16x32_bf16(false, a, false, b, (short)0, c, false, false);
}

// ---------------------------------------------------------------------------
// Elementwise prep: q_in = queries + pos*sqrt(D); k_in = keys + pos*sqrt(D)
// ---------------------------------------------------------------------------
__global__ __launch_bounds__(256) void prep_kernel(
    const float* __restrict__ q, const float* __restrict__ k,
    const float* __restrict__ pos,
    float* __restrict__ qinf, bf16_t* __restrict__ qinb, bf16_t* __restrict__ kinb) {
  size_t idx = (size_t)blockIdx.x * 256 + threadIdx.x;
  float p  = pos[idx & (size_t)(TT * DD - 1)] * PE_SCALE;
  float qq = q[idx] + p;
  float kk = k[idx] + p;
  qinf[idx] = qq;
  qinb[idx] = (bf16_t)qq;
  kinb[idx] = (bf16_t)kk;
}

// W[K][N] fp32 -> WT[N][K] bf16 (weights are small; simple version)
__global__ __launch_bounds__(256) void transpose_bf16_kernel(
    const float* __restrict__ W, bf16_t* __restrict__ WT, int K, int N) {
  int idx = blockIdx.x * 256 + threadIdx.x;
  if (idx < K * N) {
    int kk = idx / N, nn = idx - kk * N;
    WT[(size_t)nn * K + kk] = (bf16_t)W[idx];
  }
}

// v_bf[b, t, h*64+d] -> vt[(b*HH+h)*64 + d][t]  (coalesced tiled transpose)
__global__ __launch_bounds__(256) void vtrans_kernel(
    const bf16_t* __restrict__ v_bf, bf16_t* __restrict__ vt) {
  __shared__ bf16_t tile[64][65];
  const int tb = blockIdx.x;            // t-tile (64 wide)
  const int h  = blockIdx.y;
  const int b  = blockIdx.z;
  const int tx = threadIdx.x & 63;
  const int ty = threadIdx.x >> 6;      // 0..3
#pragma unroll
  for (int r = ty; r < 64; r += 4)
    tile[r][tx] = v_bf[((size_t)(b * TT + tb * 64 + r)) * DD + h * DHH + tx];
  __syncthreads();
#pragma unroll
  for (int r = ty; r < 64; r += 4)
    vt[(((size_t)(b * HH + h)) * DHH + r) * TT + tb * 64 + tx] = tile[tx][r];
}

// ---------------------------------------------------------------------------
// bf16 GEMM: C[M,N] = A[M,K] x BT[N,K]^T, fp32 accumulate.
// 256 threads / 8 waves; 128x128 macro-tile; wave = 2x4 grid of 16x16 tiles
// (32x64 per wave -> ~21 flop per L2 byte).
// Epilogue: optional residual add (fp32), optional relu, fp32 and/or bf16 out.
// ---------------------------------------------------------------------------
__global__ __launch_bounds__(256) void gemm_bf16_kernel(
    const bf16_t* __restrict__ A, const bf16_t* __restrict__ BT,
    float* __restrict__ Cf, bf16_t* __restrict__ Cbf,
    const float* __restrict__ resid, int M, int N, int K, int do_relu) {
  const int wid  = threadIdx.x >> 5;
  const int lane = threadIdx.x & 31;
  const int wm = wid & 3;
  const int wn = wid >> 2;
  const int m0 = blockIdx.x * 128 + wm * 32;
  const int n0 = blockIdx.y * 128 + wn * 64;

  v8f acc[2][4] = {};
  for (int k0 = 0; k0 < K; k0 += 32) {
    v16bf a0 = load_frag_a(A + (size_t)m0 * K + k0, K);
    v16bf a1 = load_frag_a(A + (size_t)(m0 + 16) * K + k0, K);
    v16bf bf0 = load_frag_b(BT + (size_t)(n0 +  0) * K + k0, K);
    v16bf bf1 = load_frag_b(BT + (size_t)(n0 + 16) * K + k0, K);
    v16bf bf2 = load_frag_b(BT + (size_t)(n0 + 32) * K + k0, K);
    v16bf bf3 = load_frag_b(BT + (size_t)(n0 + 48) * K + k0, K);
    acc[0][0] = wmma_bf16(a0, bf0, acc[0][0]);
    acc[0][1] = wmma_bf16(a0, bf1, acc[0][1]);
    acc[0][2] = wmma_bf16(a0, bf2, acc[0][2]);
    acc[0][3] = wmma_bf16(a0, bf3, acc[0][3]);
    acc[1][0] = wmma_bf16(a1, bf0, acc[1][0]);
    acc[1][1] = wmma_bf16(a1, bf1, acc[1][1]);
    acc[1][2] = wmma_bf16(a1, bf2, acc[1][2]);
    acc[1][3] = wmma_bf16(a1, bf3, acc[1][3]);
  }
  const int cr = lane >> 4, cc = lane & 15;
#pragma unroll
  for (int mt = 0; mt < 2; ++mt) {
#pragma unroll
    for (int j = 0; j < 4; ++j) {
      v8f a = acc[mt][j];
#pragma unroll
      for (int r = 0; r < 8; ++r) {
        int row = m0 + mt * 16 + r + cr * 8;
        int col = n0 + j * 16 + cc;
        size_t gi = (size_t)row * N + col;
        float v = a[r];
        if (resid)   v += resid[gi];
        if (do_relu) v  = fmaxf(v, 0.0f);
        if (Cf)  Cf[gi]  = v;
        if (Cbf) Cbf[gi] = (bf16_t)v;
      }
    }
  }
}

// ---------------------------------------------------------------------------
// Attention: one WG per (query-block-of-16, head, batch).
// Full 16x1024 score strip in LDS (64KB, XOR-swizzled). QK^T and PV via WMMA.
// ---------------------------------------------------------------------------
__global__ __launch_bounds__(256) void attn_kernel(
    const bf16_t* __restrict__ Q, const bf16_t* __restrict__ Kb,
    const bf16_t* __restrict__ Vt, const float* __restrict__ q_in,
    const float* __restrict__ key_masks, const float* __restrict__ query_masks,
    float* __restrict__ res_f, bf16_t* __restrict__ res_bf) {
  __shared__ float S[16 * TT];  // 64 KB
  const int wid  = threadIdx.x >> 5;
  const int lane = threadIdx.x & 31;
  const int qb = blockIdx.x * 16;
  const int h  = blockIdx.y;
  const int b  = blockIdx.z;
  const int cr = lane >> 4, cc = lane & 15;

  // ---- scores = Q K^T / 8, masked ----
  const bf16_t* Qbase = Q + ((size_t)(b * TT + qb)) * DD + h * DHH;
  v16bf aq0 = load_frag_a(Qbase, DD);
  v16bf aq1 = load_frag_a(Qbase + 32, DD);

  for (int nt = 0; nt < 8; ++nt) {
    int n0 = (wid * 8 + nt) * 16;
    const bf16_t* Kbase = Kb + ((size_t)(b * TT + n0)) * DD + h * DHH;
    v16bf bk0 = load_frag_b(Kbase, DD);
    v16bf bk1 = load_frag_b(Kbase + 32, DD);
    v8f s = {};
    s = wmma_bf16(aq0, bk0, s);
    s = wmma_bf16(aq1, bk1, s);
#pragma unroll
    for (int r = 0; r < 8; ++r) {
      int m = r + cr * 8;
      int n = n0 + cc;
      float v = s[r] * 0.125f;
      if (key_masks[b * TT + n] != 1.0f) v = NEG_BIG_F;
      if (n == qb + m)                   v = NEG_BIG_F;
      S[m * TT + SWZ(m, n)] = v;
    }
  }
  __syncthreads();

  // ---- row softmax (16 lanes per row, all within one wave) * query_mask ----
  // Swizzle is an intra-row permutation -> order-independent max/sum/scale OK.
  {
    const int row = threadIdx.x >> 4;
    const int sub = threadIdx.x & 15;
    float* srow = &S[row * TT];
    float mx = -3.4e38f;
    for (int i = sub; i < TT; i += 16) mx = fmaxf(mx, srow[i]);
#pragma unroll
    for (int o = 8; o >= 1; o >>= 1) mx = fmaxf(mx, __shfl_xor(mx, o, 32));
    float sum = 0.f;
    for (int i = sub; i < TT; i += 16) {
      float e = __expf(srow[i] - mx);
      srow[i] = e;
      sum += e;
    }
#pragma unroll
    for (int o = 8; o >= 1; o >>= 1) sum += __shfl_xor(sum, o, 32);
    float scl = query_masks[b * TT + qb + row] / sum;
    for (int i = sub; i < TT; i += 16) srow[i] *= scl;
  }
  __syncthreads();

  // ---- out = attn @ V ; each wave owns a 128-key slice of K dimension ----
  v8f o0 = {}, o1 = {}, o2 = {}, o3 = {};
  const int am = lane & 15;
  const int kh = lane >> 4;
  const bf16_t* vtb = Vt + ((size_t)(b * HH + h)) * DHH * TT;
  for (int kk = 0; kk < 128; kk += 32) {
    const int k0 = wid * 128 + kk;
    v16bf a;
    const float* p0 = &S[am * TT + SWZ(am, k0 + kh * 8)];
    const float* p1 = &S[am * TT + SWZ(am, k0 + 16 + kh * 8)];
#pragma unroll
    for (int i = 0; i < 8; ++i) { a[i] = (bf16_t)p0[i]; a[8 + i] = (bf16_t)p1[i]; }
    v8f* outs[4] = {&o0, &o1, &o2, &o3};
#pragma unroll
    for (int nt = 0; nt < 4; ++nt) {
      v16bf bv = load_frag_b(vtb + (size_t)(nt * 16) * TT + k0, TT);
      *outs[nt] = wmma_bf16(a, bv, *outs[nt]);
    }
  }
  __syncthreads();  // everyone done reading S -> safe to reuse as reduce buffer

  float* red = S;
  {
    v8f outs[4] = {o0, o1, o2, o3};
#pragma unroll
    for (int nt = 0; nt < 4; ++nt) {
      v8f a = outs[nt];
#pragma unroll
      for (int r = 0; r < 8; ++r) {
        int mm  = r + cr * 8;
        int col = nt * 16 + cc;
        red[wid * 1024 + mm * 64 + col] = a[r];
      }
    }
  }
  __syncthreads();

  for (int e = threadIdx.x; e < 1024; e += 256) {
    float acc = 0.f;
#pragma unroll
    for (int w = 0; w < 8; ++w) acc += red[w * 1024 + e];
    int mm = e >> 6, col = e & 63;
    size_t gi = ((size_t)(b * TT + qb + mm)) * DD + h * DHH + col;
    float v = acc + q_in[gi];   // residual with post-PE queries
    res_f[gi]  = v;
    res_bf[gi] = (bf16_t)v;
  }
}

// mean over T
__global__ __launch_bounds__(256) void mean_kernel(
    const float* __restrict__ fin, float* __restrict__ out) {
  int idx = blockIdx.x * 256 + threadIdx.x;  // 0 .. B*D-1
  int b = idx >> 9;
  int d = idx & 511;
  float s = 0.f;
  for (int t = 0; t < TT; ++t) s += fin[((size_t)(b * TT + t)) * DD + d];
  out[idx] = s * (1.0f / TT);
}

// ---------------------------------------------------------------------------
extern "C" void kernel_launch(void* const* d_in, const int* in_sizes, int n_in,
                              void* d_out, int out_size, void* d_ws, size_t ws_size,
                              hipStream_t stream) {
  const float* queries = (const float*)d_in[0];
  const float* keys    = (const float*)d_in[1];
  const float* qmask   = (const float*)d_in[2];
  const float* kmask   = (const float*)d_in[3];
  const float* pos     = (const float*)d_in[4];
  const float* WQ      = (const float*)d_in[5];
  const float* WK      = (const float*)d_in[6];
  const float* WV      = (const float*)d_in[7];
  const float* FW1     = (const float*)d_in[8];
  const float* FW2     = (const float*)d_in[9];
  float* out = (float*)d_out;

  char* ws = (char*)d_ws;
  const size_t MB = 1u << 20;
  float*  q_in_f  = (float*) (ws + 0);        // 16 MB
  float*  res_f   = (float*) (ws + 16 * MB);  // 16 MB
  bf16_t* q_in_bf = (bf16_t*)(ws + 32 * MB);  // 8 MB
  bf16_t* k_in_bf = (bf16_t*)(ws + 40 * MB);  // 8 MB
  bf16_t* q_bf    = (bf16_t*)(ws + 48 * MB);  // 8 MB
  bf16_t* k_bf    = (bf16_t*)(ws + 56 * MB);  // 8 MB
  bf16_t* v_bf    = (bf16_t*)(ws + 64 * MB);  // 8 MB
  bf16_t* res_bf  = (bf16_t*)(ws + 72 * MB);  // 8 MB
  bf16_t* wqt     = (bf16_t*)(ws + 80 * MB);
  bf16_t* wkt     = (bf16_t*)(ws + 80 * MB + 512 * 1024);
  bf16_t* wvt     = (bf16_t*)(ws + 81 * MB);
  bf16_t* fw1t    = (bf16_t*)(ws + 81 * MB + 512 * 1024);  // 2 MB
  bf16_t* fw2t    = (bf16_t*)(ws + 83 * MB + 512 * 1024);  // 2 MB
  bf16_t* vt_bf   = (bf16_t*)(ws + 86 * MB);  // 8 MB, V transposed per (b,h)
  bf16_t* hid_bf  = (bf16_t*)(ws + 32 * MB);  // 32 MB, reuse (dead after attn)
  float*  fin_f   = (float*) (ws + 0);        // reuse q_in_f (dead after attn)

  const int M = BB * TT;  // 8192

  prep_kernel<<<(M * DD) / 256, 256, 0, stream>>>(queries, keys, pos, q_in_f, q_in_bf, k_in_bf);

  transpose_bf16_kernel<<<(DD * DD) / 256, 256, 0, stream>>>(WQ, wqt, DD, DD);
  transpose_bf16_kernel<<<(DD * DD) / 256, 256, 0, stream>>>(WK, wkt, DD, DD);
  transpose_bf16_kernel<<<(DD * DD) / 256, 256, 0, stream>>>(WV, wvt, DD, DD);
  transpose_bf16_kernel<<<(DD * DFF) / 256, 256, 0, stream>>>(FW1, fw1t, DD, DFF);
  transpose_bf16_kernel<<<(DFF * DD) / 256, 256, 0, stream>>>(FW2, fw2t, DFF, DD);

  dim3 gp(M / 128, DD / 128);
  gemm_bf16_kernel<<<gp, 256, 0, stream>>>(q_in_bf, wqt, nullptr, q_bf, nullptr, M, DD, DD, 0);
  gemm_bf16_kernel<<<gp, 256, 0, stream>>>(k_in_bf, wkt, nullptr, k_bf, nullptr, M, DD, DD, 0);
  gemm_bf16_kernel<<<gp, 256, 0, stream>>>(k_bf,    wvt, nullptr, v_bf, nullptr, M, DD, DD, 0);

  dim3 gv(TT / 64, HH, BB);
  vtrans_kernel<<<gv, 256, 0, stream>>>(v_bf, vt_bf);

  dim3 ga(TT / 16, HH, BB);
  attn_kernel<<<ga, 256, 0, stream>>>(q_bf, k_bf, vt_bf, q_in_f, kmask, qmask, res_f, res_bf);

  dim3 g4(M / 128, DFF / 128);
  gemm_bf16_kernel<<<g4, 256, 0, stream>>>(res_bf, fw1t, nullptr, hid_bf, nullptr, M, DFF, DD, 1);
  dim3 g5(M / 128, DD / 128);
  gemm_bf16_kernel<<<g5, 256, 0, stream>>>(hid_bf, fw2t, fin_f, nullptr, res_f, M, DD, DFF, 0);

  mean_kernel<<<(BB * DD) / 256, 256, 0, stream>>>(fin_f, out);
}
